// _xLSTMBlock_65670049956059
// MI455X (gfx1250) — matrix operand order repaired and checked by reading, not compiled, not run on hardware
//
#include <hip/hip_runtime.h>
#include <hip/hip_bf16.h>

typedef __bf16 bf16;
typedef __attribute__((ext_vector_type(16))) __bf16 v16bf;
typedef __attribute__((ext_vector_type(8)))  __bf16 v8bf;
typedef __attribute__((ext_vector_type(8)))  float  v8f;

#define NWG    64
#define BLK    256
#define TSTEPS 512
#define DIN    512
#define DHID   1024
#define BATCH  64

__device__ __forceinline__ float fast_tanh(float x){
  float e = __expf(2.f * x);
  return (e - 1.f) / (e + 1.f);
}
__device__ __forceinline__ float fast_sig(float x){
  return 1.f / (1.f + __expf(-x));
}

// Device-wide barrier: one slot per step, zeroed each launch by init_ws.
__device__ __forceinline__ void gbar(unsigned* slot){
  __threadfence();
  __syncthreads();
  if (threadIdx.x == 0){
    __hip_atomic_fetch_add(slot, 1u, __ATOMIC_ACQ_REL, __HIP_MEMORY_SCOPE_AGENT);
    while (__hip_atomic_load(slot, __ATOMIC_ACQUIRE, __HIP_MEMORY_SCOPE_AGENT) < (unsigned)NWG){
      __builtin_amdgcn_s_sleep(1);
    }
  }
  __syncthreads();
  __threadfence();
}

// A-matrix 16x32 bf16, contiguous k order (matches B):
// lanes 0-15 : M=lane,   elems 0..15 -> K=kb+0..15
// lanes 16-31: M=lane-16,elems 0..15 -> K=kb+16..31
__device__ __forceinline__ v16bf load_a(const bf16* row, int kb, int lane){
  int koff = kb + ((lane >> 4) << 4);
  v8bf lo = *(const v8bf*)(row + koff);
  v8bf hi = *(const v8bf*)(row + koff + 8);
  v16bf a;
#pragma unroll
  for (int e = 0; e < 8; ++e){ a[e] = lo[e]; a[e + 8] = hi[e]; }
  return a;
}

// One-time: combined x-side weights (W_xc ; W_xg) -> bf16, row-major 4096x512.
__global__ void prep_wx(const float* __restrict__ W_xc, const float* __restrict__ W_xg,
                        bf16* __restrict__ wxq){
  int i = blockIdx.x * blockDim.x + threadIdx.x;
  if (i < 4096 * DIN){
    int r = i >> 9;
    int c = i & (DIN - 1);
    float v = (r < DHID) ? W_xc[(size_t)r * DIN + c]
                         : W_xg[(size_t)(r - DHID) * DIN + c];
    wxq[i] = (bf16)v;
  }
}

// One-time, fully parallel: LayerNorm every x row (all b, all t) -> bf16.
__global__ __launch_bounds__(BLK) void ln_x_all(const float* __restrict__ x,
    const float* __restrict__ in_w, const float* __restrict__ in_b,
    bf16* __restrict__ xn_all){
  __shared__ float rA[BLK], rB[BLK];
  const int row = blockIdx.x;              // b*TSTEPS + t
  const int tid = threadIdx.x;
  const float* xr = x + (size_t)row * DIN;
  float s = 0.f, ss = 0.f;
  for (int k = tid; k < DIN; k += BLK){ float v = xr[k]; s += v; ss += v * v; }
  rA[tid] = s; rB[tid] = ss; __syncthreads();
  for (int off = BLK >> 1; off > 0; off >>= 1){
    if (tid < off){ rA[tid] += rA[tid + off]; rB[tid] += rB[tid + off]; }
    __syncthreads();
  }
  float mu   = rA[0] * (1.f / DIN);
  float var  = rB[0] * (1.f / DIN) - mu * mu;
  float rstd = __frsqrt_rn(var + 1e-5f);
  bf16* orow = xn_all + (size_t)row * DIN;
  for (int k = tid; k < DIN; k += BLK){
    float v = xr[k];
    orow[k] = (bf16)((v - mu) * rstd * in_w[k] + in_b[k]);
  }
}

__global__ void init_ws(bf16* __restrict__ hraw, float* __restrict__ statsG,
                        unsigned* __restrict__ bars){
  int i = blockIdx.x * blockDim.x + threadIdx.x;
  if (i < 2 * BATCH * DHID) hraw[i] = (bf16)0.f;
  if (i < (TSTEPS + 1) * BATCH * 2) statsG[i] = 0.f;
  if (i < TSTEPS) bars[i] = 0u;
}

__global__ __launch_bounds__(BLK) void xlstm_persistent(
    const int*   __restrict__ lengths,
    const float* __restrict__ h_w,  const float* __restrict__ h_b,
    const float* __restrict__ b_xc, const float* __restrict__ b_xg,
    const float* __restrict__ W_hc, const float* __restrict__ W_hg,
    const bf16*  __restrict__ wxq,      // 4096 x 512 bf16
    const bf16*  __restrict__ xn_all,   // 64 x 512 x 512 bf16 (LN'd x)
    bf16*     __restrict__ hraw,        // 2 x 64 x 1024 bf16 (raw h, double-buffered)
    float*    __restrict__ statsG,      // (TSTEPS+1) x 64 x 2 f32 (sum, sumsq of h)
    unsigned* __restrict__ bars,        // TSTEPS barrier slots
    float*    __restrict__ out)         // 64 x 512 x 1024
{
  __shared__ alignas(32) bf16  sWh[64 * DHID];       // 128 KB : w-folded W_h strip
  __shared__ alignas(32) float sStage[4 * 64 * 16];  // 16 KB
  __shared__ float sC[64 * 16];                      // 4 KB
  __shared__ float sRedA[BLK], sRedB[BLK];           // 2 KB
  __shared__ float sBias[64];                        // bias + sum_k b[k]*Wh[n,k]
  __shared__ float sS[64];                           // S[n] = sum_k w[k]*Wh[n,k] (bf16-exact)
  __shared__ float sMu[BATCH], sRstd[BATCH];

  const int g   = blockIdx.x;      // owns h-columns [g*16, g*16+16)
  const int tid = threadIdx.x;

  // ---- prologue 1: cache w-folded W_h strip (cell,i,f,o rows) as bf16 ----
  for (int i = tid; i < 64 * DHID; i += BLK){
    int lr = i >> 10;              // local row = gt*16 + jj
    int k  = i & (DHID - 1);
    int gt = lr >> 4, jj = lr & 15;
    int j  = g * 16 + jj;
    float v = (gt == 0) ? W_hc[(size_t)j * DHID + k]
                        : W_hg[(size_t)((gt - 1) * DHID + j) * DHID + k];
    sWh[i] = (bf16)(v * h_w[k]);   // fold LN scale into weights
  }
  for (int i = tid; i < 64 * 16; i += BLK) sC[i] = 0.f;
  __syncthreads();

  // ---- prologue 2: per-row S[n] (from quantized weights) and fused bias ----
  {
    int lr = tid >> 2;             // 4 threads per local row
    int part = tid & 3;
    int gt = lr >> 4, jj = lr & 15;
    int j  = g * 16 + jj;
    const float* wrow = (gt == 0) ? (W_hc + (size_t)j * DHID)
                                  : (W_hg + (size_t)((gt - 1) * DHID + j) * DHID);
    float aS = 0.f, aB = 0.f;
    for (int k = part * (DHID / 4); k < (part + 1) * (DHID / 4); ++k){
      aS += (float)sWh[lr * DHID + k];
      aB += h_b[k] * wrow[k];      // LN shift term folded into bias
    }
    sRedA[tid] = aS; sRedB[tid] = aB;
  }
  __syncthreads();
  if (tid < 64){
    float S  = sRedA[tid * 4] + sRedA[tid * 4 + 1] + sRedA[tid * 4 + 2] + sRedA[tid * 4 + 3];
    float Bb = sRedB[tid * 4] + sRedB[tid * 4 + 1] + sRedB[tid * 4 + 2] + sRedB[tid * 4 + 3];
    int gt = tid >> 4, jj = tid & 15;
    int j  = g * 16 + jj;
    float ob = (gt == 0) ? b_xc[j] : b_xg[(gt - 1) * DHID + j];
    sS[tid]    = S;
    sBias[tid] = ob + Bb;
  }
  __syncthreads();

  const int lane = tid & 31;
  const int wid  = tid >> 5;
  const int mt   = wid >> 1;
  const int gt0  = (wid & 1) << 1;
  const int nn   = lane & 15;
  const int bko  = (lane >> 4) << 4;
  const int mrow = mt * 16 + nn;

  const bf16* bx0 = wxq + ((size_t)(gt0    ) * DHID + g * 16 + nn) * DIN + bko;
  const bf16* bx1 = wxq + ((size_t)(gt0 + 1) * DHID + g * 16 + nn) * DIN + bko;
  const bf16* bh0 = sWh + ((gt0    ) * 16 + nn) * DHID + bko;
  const bf16* bh1 = sWh + ((gt0 + 1) * 16 + nn) * DHID + bko;

  for (int t = 0; t < TSTEPS; ++t){
    // ---- per-step mu/rstd from accumulated stats (slot t) ----
    if (tid < BATCH){
      float s   = statsG[((size_t)t * BATCH + tid) * 2 + 0];
      float ss  = statsG[((size_t)t * BATCH + tid) * 2 + 1];
      float mu  = s * (1.f / DHID);
      float var = ss * (1.f / DHID) - mu * mu;
      sMu[tid]   = mu;
      sRstd[tid] = __frsqrt_rn(var + 1e-5f);
    }
    __syncthreads();

    // ---- GEMM: x-part (L2 weights) + raw-h part (LDS weights), separate accs ----
    v8f accX0 = {}; v8f accX1 = {};
    v8f accH0 = {}; v8f accH1 = {};
    {
      const bf16* arow = xn_all + ((size_t)mrow * TSTEPS + t) * DIN;
#pragma unroll 4
      for (int kb = 0; kb < DIN; kb += 32){
        v16bf a  = load_a(arow, kb, lane);
        v16bf b0 = *(const v16bf*)(bx0 + kb);
        v16bf b1 = *(const v16bf*)(bx1 + kb);
        accX0 = __builtin_amdgcn_wmma_f32_16x16x32_bf16(false, a, false, b0, (short)0, accX0, false, false);
        accX1 = __builtin_amdgcn_wmma_f32_16x16x32_bf16(false, a, false, b1, (short)0, accX1, false, false);
      }
      const bf16* ahr = hraw + (size_t)(t & 1) * BATCH * DHID + (size_t)mrow * DHID;
#pragma unroll 4
      for (int kb = 0; kb < DHID; kb += 32){
        v16bf a  = load_a(ahr, kb, lane);
        v16bf b0 = *(const v16bf*)(bh0 + kb);
        v16bf b1 = *(const v16bf*)(bh1 + kb);
        accH0 = __builtin_amdgcn_wmma_f32_16x16x32_bf16(false, a, false, b0, (short)0, accH0, false, false);
        accH1 = __builtin_amdgcn_wmma_f32_16x16x32_bf16(false, a, false, b1, (short)0, accH1, false, false);
      }
    }

    // ---- epilogue: apply folded LayerNorm + bias, stage gate pre-activations ----
    {
      float Sn0 = sS[(gt0    ) * 16 + nn], B0 = sBias[(gt0    ) * 16 + nn];
      float Sn1 = sS[(gt0 + 1) * 16 + nn], B1 = sBias[(gt0 + 1) * 16 + nn];
      int mb = (lane >> 4) << 3;
#pragma unroll
      for (int v = 0; v < 8; ++v){
        int m = mt * 16 + mb + v;
        float mu = sMu[m], rs = sRstd[m];
        sStage[(gt0    ) * 1024 + m * 16 + nn] = accX0[v] + B0 + rs * (accH0[v] - mu * Sn0);
        sStage[(gt0 + 1) * 1024 + m * 16 + nn] = accX1[v] + B1 + rs * (accH1[v] - mu * Sn1);
      }
    }
    __syncthreads();

    // ---- elementwise gate math, state update, stats for next step ----
    bf16* hnext = hraw + (size_t)((t + 1) & 1) * BATCH * DHID;
    const bf16* hcur = hraw + (size_t)(t & 1) * BATCH * DHID;
    float s4 = 0.f, ss4 = 0.f;
#pragma unroll
    for (int q = 0; q < 4; ++q){
      int idx = tid * 4 + q;            // (m, jj); 4 consecutive share m
      int m  = idx >> 4;
      int jj = idx & 15;
      int j  = g * 16 + jj;
      float cell = sStage[0 * 1024 + idx];
      float gi   = sStage[1 * 1024 + idx];
      float gf   = sStage[2 * 1024 + idx];
      float go   = sStage[3 * 1024 + idx];
      float gg = fast_tanh(cell);
      float ih = __expf(fminf(fmaxf(gi, -5.f), 5.f));
      float fh = __expf(fminf(fmaxf(gf, -5.f), 5.f));
      float denom = ih + fh + 1e-6f;
      float c_old = sC[idx];
      float c_new = (fh * c_old + ih * gg) / denom;
      float h_new = fast_sig(go) * fast_tanh(c_new);
      bool  valid = t < lengths[m];
      float h_old = (float)hcur[m * DHID + j];
      float h2 = valid ? h_new : h_old;
      sC[idx] = valid ? c_new : c_old;
      hnext[m * DHID + j] = (bf16)h2;
      out[((size_t)m * TSTEPS + t) * DHID + j] = valid ? h_new : 0.f;
      s4 += h2; ss4 += h2 * h2;
    }
    sRedA[tid] = s4; sRedB[tid] = ss4;
    __syncthreads();
    if (tid < BATCH){                    // one thread per batch row: 2 atomics
      float s  = sRedA[tid * 4] + sRedA[tid * 4 + 1] + sRedA[tid * 4 + 2] + sRedA[tid * 4 + 3];
      float ss = sRedB[tid * 4] + sRedB[tid * 4 + 1] + sRedB[tid * 4 + 2] + sRedB[tid * 4 + 3];
      atomicAdd(&statsG[((size_t)(t + 1) * BATCH + tid) * 2 + 0], s);
      atomicAdd(&statsG[((size_t)(t + 1) * BATCH + tid) * 2 + 1], ss);
    }
    gbar(&bars[t]);                      // single device-wide barrier per step
  }
}

extern "C" void kernel_launch(void* const* d_in, const int* in_sizes, int n_in,
                              void* d_out, int out_size, void* d_ws, size_t ws_size,
                              hipStream_t stream) {
  const float* x    = (const float*)d_in[0];
  const int*   lens = (const int*)  d_in[1];
  const float* inw  = (const float*)d_in[2];
  const float* inb  = (const float*)d_in[3];
  const float* hw   = (const float*)d_in[4];
  const float* hb   = (const float*)d_in[5];
  const float* W_xc = (const float*)d_in[6];
  const float* bxc  = (const float*)d_in[7];
  const float* W_hc = (const float*)d_in[8];
  const float* W_xg = (const float*)d_in[9];
  const float* bxg  = (const float*)d_in[10];
  const float* W_hg = (const float*)d_in[11];

  char* ws = (char*)d_ws;
  bf16*     hraw   = (bf16*)    (ws + 0);             // 256 KB (2 buffers)
  float*    statsG = (float*)   (ws + (256  << 10));  // 513*64*2 f32 ~ 257 KB
  unsigned* bars   = (unsigned*)(ws + (768  << 10));  // 2 KB
  bf16*     wxq    = (bf16*)    (ws + (832  << 10));  // 4 MB
  bf16*     xn_all = (bf16*)    (ws + (5120 << 10));  // 32 MB

  prep_wx<<<(4096 * DIN + 255) / 256, 256, 0, stream>>>(W_xc, W_xg, wxq);
  ln_x_all<<<BATCH * TSTEPS, BLK, 0, stream>>>(x, inw, inb, xn_all);
  init_ws<<<(2 * BATCH * DHID + 255) / 256, 256, 0, stream>>>(hraw, statsG, bars);
  xlstm_persistent<<<NWG, BLK, 0, stream>>>(
      lens, hw, hb, bxc, bxg, W_hc, W_hg,
      wxq, xn_all, hraw, statsG, bars, (float*)d_out);
}
